// PointTransformer_seg_8581344658090
// MI455X (gfx1250) — compile-verified, matrix operand order checked by reading, not run on hardware
//
#include <hip/hip_runtime.h>
#include <hip/hip_bf16.h>

// ---------------- model dims (fixed) ----------------
#define BB   16
#define NN   2048
#define GG   128
#define MM   32
#define DD   384
#define HH   6
#define DHh  64
#define ENCC 256
#define NCLS 50
#define EPSV 1e-5f
#define BN_INV 0.99999500003749969f   // 1/sqrt(1+1e-5)

typedef _Float16 h16;
typedef __attribute__((ext_vector_type(8)))  _Float16 v8h;
typedef __attribute__((ext_vector_type(16))) _Float16 v16h;
typedef __attribute__((ext_vector_type(8)))  float    v8f;

static __device__ __forceinline__ v8h pack8(float4 a, float4 b)
{
  union { unsigned u[4]; v8h h; } r;
  r.u[0] = __builtin_bit_cast(unsigned, __builtin_amdgcn_cvt_pkrtz(a.x, a.y));
  r.u[1] = __builtin_bit_cast(unsigned, __builtin_amdgcn_cvt_pkrtz(a.z, a.w));
  r.u[2] = __builtin_bit_cast(unsigned, __builtin_amdgcn_cvt_pkrtz(b.x, b.y));
  r.u[3] = __builtin_bit_cast(unsigned, __builtin_amdgcn_cvt_pkrtz(b.z, b.w));
  return r.h;
}

// =====================================================================
// Generic WMMA GEMM:  C[M,N] = epi( A[M,K] @ W[N,K]^T + bias + res )
//   epi: optional per-channel bn (x*BN_INV*g + b), act: 0=none,1=relu,2=gelu
// Block = 128 threads (4 waves). Tile: 64x64 C, K-step 32.
// Fast path: b128 global loads -> v_cvt_pk_rtz -> ds_store_b128.
// Edge path: branchless clamped-index scalar loads.
// =====================================================================
__global__ __launch_bounds__(128) void k_gemm_wmma(
    const float* __restrict__ A, const float* __restrict__ W,
    const float* __restrict__ bias, const float* __restrict__ bn_g,
    const float* __restrict__ bn_b, const float* __restrict__ res,
    float* __restrict__ C, int Mrows, int K, int Nc, int act)
{
  __shared__ h16 As[64 * 32];
  __shared__ h16 Ws[64 * 32];
  const int tid  = threadIdx.x;
  const int wave = tid >> 5;
  const int lane = tid & 31;
  const int m0 = blockIdx.y * 64;
  const int n0 = blockIdx.x * 64;

  v8f acc[4];
#pragma unroll
  for (int i = 0; i < 4; ++i) { v8f z = {}; acc[i] = z; }

  const bool mfull = (m0 + 64 <= Mrows);
  const bool nfull = (n0 + 64 <= Nc);
  const bool k4    = ((K & 3) == 0);
  const int  lr = tid >> 1;             // 0..63: tile row handled by this thread
  const int  lh = (tid & 1) << 4;       // 0 / 16: half-row (16 cols)

  for (int k0 = 0; k0 < K; k0 += 32) {
    const bool kfull = (k0 + 32 <= K);

    if (mfull & kfull & k4) {           // vector path for A
      const float* ap = A + (size_t)(m0 + lr) * K + (k0 + lh);
      float4 a0 = *(const float4*)(ap);
      float4 a1 = *(const float4*)(ap + 4);
      float4 a2 = *(const float4*)(ap + 8);
      float4 a3 = *(const float4*)(ap + 12);
      *(v8h*)(As + (lr << 5) + lh)     = pack8(a0, a1);
      *(v8h*)(As + (lr << 5) + lh + 8) = pack8(a2, a3);
    } else {
#pragma unroll 4
      for (int i = tid; i < 64 * 32; i += 128) {
        int rr = i >> 5, cc = i & 31;
        int gr = m0 + rr, gc = k0 + cc;
        bool ok = (gr < Mrows) & (gc < K);
        size_t ix = ok ? ((size_t)gr * K + gc) : 0;
        float av = A[ix];
        As[(rr << 5) + cc] = (h16)(ok ? av : 0.f);
      }
    }
    if (nfull & kfull & k4) {           // vector path for W
      const float* wp = W + (size_t)(n0 + lr) * K + (k0 + lh);
      float4 w0 = *(const float4*)(wp);
      float4 w1 = *(const float4*)(wp + 4);
      float4 w2 = *(const float4*)(wp + 8);
      float4 w3 = *(const float4*)(wp + 12);
      *(v8h*)(Ws + (lr << 5) + lh)     = pack8(w0, w1);
      *(v8h*)(Ws + (lr << 5) + lh + 8) = pack8(w2, w3);
    } else {
#pragma unroll 4
      for (int i = tid; i < 64 * 32; i += 128) {
        int rr = i >> 5, cc = i & 31;
        int nr = n0 + rr, gc = k0 + cc;
        bool ok = (nr < Nc) & (gc < K);
        size_t ix = ok ? ((size_t)nr * K + gc) : 0;
        float wv = W[ix];
        Ws[(rr << 5) + cc] = (h16)(ok ? wv : 0.f);
      }
    }
    __syncthreads();

    // prefetch next K-slab of A (global_prefetch_b8)
    if (k0 + 32 < K) {
      int gr = m0 + (tid & 63);
      if (gr < Mrows) __builtin_prefetch(&A[(size_t)gr * K + k0 + 32], 0, 3);
    }

    const int arow = (wave << 4) + (lane & 15);
    const int koff = (lane >> 4) << 4;           // lanes 0-15: K 0..15, 16-31: K 16..31
    v16h af = *(const v16h*)(As + (arow << 5) + koff);
    v16h wf[4];
#pragma unroll
    for (int nt = 0; nt < 4; ++nt)
      wf[nt] = *(const v16h*)(Ws + ((((nt << 4) + (lane & 15)) << 5) + koff));
#pragma unroll
    for (int nt = 0; nt < 4; ++nt)
      acc[nt] = __builtin_amdgcn_wmma_f32_16x16x32_f16(
          false, af, false, wf[nt], (short)0, acc[nt], false, false);
    __syncthreads();
  }

  // epilogue; C/D layout: VGPR r holds M = r + (lane<16?0:8), N = lane&15
  const int mbase = m0 + (wave << 4) + ((lane >> 4) << 3);
  const int nbase = n0 + (lane & 15);
#pragma unroll
  for (int nt = 0; nt < 4; ++nt) {
    int n = nbase + (nt << 4);
    if (n >= Nc) continue;
    float bsv = bias ? bias[n] : 0.f;
    float gv  = bn_g ? bn_g[n] : 1.f;
    float bbv = bn_b ? bn_b[n] : 0.f;
#pragma unroll
    for (int r = 0; r < 8; ++r) {
      int m = mbase + r;
      if (m >= Mrows) continue;
      float v = acc[nt][r] + bsv;
      if (bn_g) v = v * BN_INV * gv + bbv;
      if (res)  v += res[(size_t)m * Nc + n];
      if (act == 1)      v = fmaxf(v, 0.f);
      else if (act == 2) v = 0.5f * v * (1.f + erff(v * 0.70710678f));
      C[(size_t)m * Nc + n] = v;
    }
  }
}

// =====================================================================
// Farthest point sampling: one block per batch, 2048 pts -> 512 coords.
// (center/cp2 = first 128, cp1 = first 256, cp0 = all 512: same scan.)
// =====================================================================
__global__ __launch_bounds__(256) void k_fps(const float* __restrict__ pts,
                                             float* __restrict__ fxyz, int nsample)
{
  const int b = blockIdx.x, tid = threadIdx.x;
  const float* P = pts + (size_t)b * NN * 3;
  float dmin[8], px[8], py[8], pz[8];
#pragma unroll
  for (int i = 0; i < 8; ++i) {
    int p = tid + i * 256;
    dmin[i] = 1e30f;
    px[i] = P[p * 3]; py[i] = P[p * 3 + 1]; pz[i] = P[p * 3 + 2];
  }
  __shared__ float sv[256];
  __shared__ int   si[256];
  __shared__ float cur[3];
  if (tid == 0) { cur[0] = P[0]; cur[1] = P[1]; cur[2] = P[2]; }
  __syncthreads();
  for (int it = 0; it < nsample; ++it) {
    float cx = cur[0], cy = cur[1], cz = cur[2];
    if (tid == 0) {
      float* o = fxyz + ((size_t)b * nsample + it) * 3;
      o[0] = cx; o[1] = cy; o[2] = cz;
    }
    float best = -1.f; int bi = 0;
#pragma unroll
    for (int i = 0; i < 8; ++i) {
      float dx = px[i] - cx, dy = py[i] - cy, dz = pz[i] - cz;
      float d = dx * dx + dy * dy + dz * dz;
      dmin[i] = fminf(dmin[i], d);
      if (dmin[i] > best) { best = dmin[i]; bi = tid + i * 256; }
    }
    sv[tid] = best; si[tid] = bi;
    __syncthreads();
    for (int s = 128; s > 0; s >>= 1) {
      if (tid < s && sv[tid + s] > sv[tid]) { sv[tid] = sv[tid + s]; si[tid] = si[tid + s]; }
      __syncthreads();
    }
    if (tid == 0) {
      int ns = si[0];
      cur[0] = P[ns * 3]; cur[1] = P[ns * 3 + 1]; cur[2] = P[ns * 3 + 2];
    }
    __syncthreads();
  }
}

// ---------------- k-NN top-k (insertion sort per query) ----------------
template <int KK>
__global__ void k_knn(const float* __restrict__ q, const float* __restrict__ kp,
                      int* __restrict__ idx, float* __restrict__ dist, int Q, int Nk)
{
  int b = blockIdx.y;
  int qi = blockIdx.x * blockDim.x + threadIdx.x;
  if (qi >= Q) return;
  const float* qp = q + ((size_t)b * Q + qi) * 3;
  float qx = qp[0], qy = qp[1], qz = qp[2];
  float bd[KK]; int bi[KK];
#pragma unroll
  for (int i = 0; i < KK; ++i) { bd[i] = 1e30f; bi[i] = 0; }
  const float* kb = kp + (size_t)b * Nk * 3;
  for (int j = 0; j < Nk; ++j) {
    float dx = kb[j * 3] - qx, dy = kb[j * 3 + 1] - qy, dz = kb[j * 3 + 2] - qz;
    float d = dx * dx + dy * dy + dz * dz;
    if (d < bd[KK - 1]) {
      int p = KK - 1;
      while (p > 0 && bd[p - 1] > d) { bd[p] = bd[p - 1]; bi[p] = bi[p - 1]; --p; }
      bd[p] = d; bi[p] = j;
    }
  }
  size_t base = ((size_t)b * Q + qi) * KK;
#pragma unroll
  for (int i = 0; i < KK; ++i) {
    idx[base + i] = bi[i];
    if (dist) dist[base + i] = bd[i];
  }
}

// ---------------- misc elementwise kernels ----------------
__global__ void k_subset3(const float* __restrict__ src, float* __restrict__ dst,
                          int srcRows, long total)   // dst[b][r][3] = src[b*srcRows + r][3]
{
  long t = (long)blockIdx.x * 256 + threadIdx.x;
  if (t >= total) return;
  long per = total / BB;                 // nRows*3
  int b = (int)(t / per); long rem = t % per;
  dst[t] = src[(size_t)b * srcRows * 3 + rem];
}

__global__ void k_gather_nb(const float* __restrict__ pts, const float* __restrict__ center,
                            const int* __restrict__ nidx, float* __restrict__ nb, long total)
{
  long t = (long)blockIdx.x * 256 + threadIdx.x;
  if (t >= total) return;
  long bg = t / MM;
  int b = (int)(bg / GG);
  int pi = nidx[t];
  const float* pp = pts + ((size_t)b * NN + pi) * 3;
  const float* cc = center + (size_t)bg * 3;
  float* o = nb + (size_t)t * 3;
  o[0] = pp[0] - cc[0]; o[1] = pp[1] - cc[1]; o[2] = pp[2] - cc[2];
}

__global__ void k_maxpool(const float* __restrict__ x, float* __restrict__ out,
                          int Mn, int C, long total)   // x[R][Mn][C] -> out[R][C]
{
  long t = (long)blockIdx.x * 256 + threadIdx.x;
  if (t >= total) return;
  long r = t / C; int c = (int)(t % C);
  float best = -1e30f;
  const float* xp = x + ((size_t)r * Mn) * C + c;
  for (int m = 0; m < Mn; ++m) best = fmaxf(best, xp[(size_t)m * C]);
  out[t] = best;
}

__global__ void k_concat_enc(const float* __restrict__ hg, const float* __restrict__ h2,
                             float* __restrict__ out, long total)  // [bc(hg) | h2], C=256
{
  long t = (long)blockIdx.x * 256 + threadIdx.x;
  if (t >= total) return;
  long row = t / 512; int c = (int)(t % 512);
  out[t] = (c < 256) ? hg[(size_t)(row / MM) * 256 + c]
                     : h2[(size_t)row * 256 + (c - 256)];
}

__global__ void k_add(const float* __restrict__ a, const float* __restrict__ b,
                      float* __restrict__ o, long total)
{
  long t = (long)blockIdx.x * 256 + threadIdx.x;
  if (t >= total) return;
  o[t] = a[t] + b[t];
}

// ---------------- LayerNorm (one block per row) ----------------
__global__ __launch_bounds__(128) void k_ln(const float* __restrict__ x,
    const float* __restrict__ g, const float* __restrict__ bta,
    float* __restrict__ y, int C)
{
  __shared__ float red[128];
  const int row = blockIdx.x, tid = threadIdx.x;
  const float* xr = x + (size_t)row * C;
  float s = 0.f;
  for (int c = tid; c < C; c += 128) s += xr[c];
  red[tid] = s; __syncthreads();
  for (int t = 64; t > 0; t >>= 1) { if (tid < t) red[tid] += red[tid + t]; __syncthreads(); }
  const float mu = red[0] / C;
  __syncthreads();
  float s2 = 0.f;
  for (int c = tid; c < C; c += 128) { float d = xr[c] - mu; s2 += d * d; }
  red[tid] = s2; __syncthreads();
  for (int t = 64; t > 0; t >>= 1) { if (tid < t) red[tid] += red[tid + t]; __syncthreads(); }
  const float inv = rsqrtf(red[0] / C + EPSV);
  float* yr = y + (size_t)row * C;
  for (int c = tid; c < C; c += 128) yr[c] = (xr[c] - mu) * inv * g[c] + bta[c];
}

// ---------------- fused attention: one block per (b,head) ----------------
// K/V tiles staged to LDS with CDNA5 async global->LDS b128 copies
// (tracked by ASYNCcnt), then float4 LDS traffic for the dot products.
__global__ __launch_bounds__(128) void k_attn(const float* __restrict__ qkv,
                                              float* __restrict__ o)
{
  extern __shared__ float smem[];           // K[128*64] | V[128*64]
  const int bh = blockIdx.x, b = bh / HH, h = bh % HH;
  const float* base = qkv + (size_t)b * GG * (3 * DD);
  float* Ks = smem;
  float* Vs = smem + GG * DHh;
#pragma unroll
  for (int j = 0; j < 16; ++j) {
    int cidx = threadIdx.x + j * 128;     // over 2048 float4 chunks of K (and V)
    int r = cidx >> 4, c4 = cidx & 15;
    const float* gk = base + (size_t)r * (3 * DD) + DD + h * DHh + c4 * 4;
    const float* gv = gk + DD;
    unsigned lk = (unsigned)(uintptr_t)(Ks + r * DHh + c4 * 4);
    unsigned lv = (unsigned)(uintptr_t)(Vs + r * DHh + c4 * 4);
    asm volatile("global_load_async_to_lds_b128 %0, %1, off"
                 :: "v"(lk), "v"(gk) : "memory");
    asm volatile("global_load_async_to_lds_b128 %0, %1, off"
                 :: "v"(lv), "v"(gv) : "memory");
  }
  asm volatile("s_wait_asynccnt 0x0" ::: "memory");
  __syncthreads();

  const int m = threadIdx.x;
  const float4* qp4 = (const float4*)(base + (size_t)m * (3 * DD) + h * DHh);
  float4 qv[16];
#pragma unroll
  for (int c4 = 0; c4 < 16; ++c4) {
    float4 v = qp4[c4];
    qv[c4] = make_float4(v.x * 0.125f, v.y * 0.125f, v.z * 0.125f, v.w * 0.125f);
  }
  const float4* K4 = (const float4*)Ks;
  const float4* V4 = (const float4*)Vs;
  float mx = -1e30f;
  for (int j = 0; j < GG; ++j) {
    float s = 0.f;
#pragma unroll
    for (int c4 = 0; c4 < 16; ++c4) {
      float4 kv = K4[j * 16 + c4];
      s += qv[c4].x * kv.x + qv[c4].y * kv.y + qv[c4].z * kv.z + qv[c4].w * kv.w;
    }
    mx = fmaxf(mx, s);
  }
  float4 av[16];
#pragma unroll
  for (int c4 = 0; c4 < 16; ++c4) av[c4] = make_float4(0.f, 0.f, 0.f, 0.f);
  float sum = 0.f;
  for (int j = 0; j < GG; ++j) {
    float s = 0.f;
#pragma unroll
    for (int c4 = 0; c4 < 16; ++c4) {
      float4 kv = K4[j * 16 + c4];
      s += qv[c4].x * kv.x + qv[c4].y * kv.y + qv[c4].z * kv.z + qv[c4].w * kv.w;
    }
    float e = expf(s - mx);
    sum += e;
#pragma unroll
    for (int c4 = 0; c4 < 16; ++c4) {
      float4 vv = V4[j * 16 + c4];
      av[c4].x += e * vv.x; av[c4].y += e * vv.y;
      av[c4].z += e * vv.z; av[c4].w += e * vv.w;
    }
  }
  float inv = 1.f / sum;
  float* op = o + ((size_t)b * GG + m) * DD + h * DHh;
#pragma unroll
  for (int c4 = 0; c4 < 16; ++c4) {
    op[c4 * 4 + 0] = av[c4].x * inv;
    op[c4 * 4 + 1] = av[c4].y * inv;
    op[c4 * 4 + 2] = av[c4].z * inv;
    op[c4 * 4 + 3] = av[c4].w * inv;
  }
}

// ---------------- feature propagation: 3-NN inverse-distance interp + concat ----------------
__global__ void k_fp_interp(const int* __restrict__ idx, const float* __restrict__ dist,
                            const float* __restrict__ kf, const float* __restrict__ extra,
                            float* __restrict__ out, int Q, int Kn, int C, int CX, long rows)
{
  long t = (long)blockIdx.x * 256 + threadIdx.x;
  if (t >= rows) return;
  int b = (int)(t / Q);
  const int*   ip = idx  + (size_t)t * 3;
  const float* dp = dist + (size_t)t * 3;
  float w0 = 1.f / (fmaxf(dp[0], 0.f) + 1e-8f);
  float w1 = 1.f / (fmaxf(dp[1], 0.f) + 1e-8f);
  float w2 = 1.f / (fmaxf(dp[2], 0.f) + 1e-8f);
  float ws = 1.f / (w0 + w1 + w2);
  w0 *= ws; w1 *= ws; w2 *= ws;
  const float* kb = kf + (size_t)b * Kn * C;
  const float* f0 = kb + (size_t)ip[0] * C;
  const float* f1 = kb + (size_t)ip[1] * C;
  const float* f2 = kb + (size_t)ip[2] * C;
  float* op = out + (size_t)t * (C + CX);
  for (int c = 0; c < C; ++c) op[c] = w0 * f0[c] + w1 * f1[c] + w2 * f2[c];
  const float* ep = extra + (size_t)t * CX;
  for (int c = 0; c < CX; ++c) op[C + c] = ep[c];
}

// ---------------- DGCNN graph feature: [feat-xq | xq] ----------------
__global__ void k_graphfeat(const int* __restrict__ idx, const float* __restrict__ xk,
                            const float* __restrict__ xq, float* __restrict__ out,
                            int Q, int Kn, int C, long total)   // total = B*Q*4
{
  long t = (long)blockIdx.x * 256 + threadIdx.x;
  if (t >= total) return;
  long bq = t / 4;
  int b = (int)(bq / Q);
  int j = idx[t];
  const float* fk = xk + ((size_t)b * Kn + j) * C;
  const float* fq = xq + (size_t)bq * C;
  float* op = out + (size_t)t * 2 * C;
  for (int c = 0; c < C; ++c) { float a = fq[c]; op[c] = fk[c] - a; op[C + c] = a; }
}

// ---------------- GroupNorm(4) stats + apply+leaky+max over k ----------------
__global__ __launch_bounds__(256) void k_gn_stats(const float* __restrict__ x,
                                                  float* __restrict__ stats, int R, int C)
{
  __shared__ float rs[256], rs2[256];
  const int b = blockIdx.y, g = blockIdx.x, Cg = C / 4;
  const float* xb = x + (size_t)b * R * C;
  float s = 0.f, s2 = 0.f;
  for (long i = threadIdx.x; i < (long)R * Cg; i += 256) {
    long r = i / Cg; int c = g * Cg + (int)(i % Cg);
    float v = xb[(size_t)r * C + c];
    s += v; s2 += v * v;
  }
  rs[threadIdx.x] = s; rs2[threadIdx.x] = s2;
  __syncthreads();
  for (int t = 128; t > 0; t >>= 1) {
    if (threadIdx.x < t) { rs[threadIdx.x] += rs[threadIdx.x + t]; rs2[threadIdx.x] += rs2[threadIdx.x + t]; }
    __syncthreads();
  }
  if (threadIdx.x == 0) {
    float cnt = (float)R * Cg;
    float mu = rs[0] / cnt;
    stats[(b * 4 + g) * 2 + 0] = mu;
    stats[(b * 4 + g) * 2 + 1] = rs2[0] / cnt - mu * mu;
  }
}

__global__ void k_gn_apply_max(const float* __restrict__ x, const float* __restrict__ stats,
                               const float* __restrict__ g, const float* __restrict__ be,
                               float* __restrict__ out, int Nq, int C, long total)
{
  long t = (long)blockIdx.x * 256 + threadIdx.x;       // over B*Nq*C
  if (t >= total) return;
  int c = (int)(t % C);
  long bn = t / C;
  int b = (int)(bn / Nq);
  int grp = c / (C / 4);
  float mu = stats[(b * 4 + grp) * 2], var = stats[(b * 4 + grp) * 2 + 1];
  float inv = rsqrtf(var + EPSV);
  float gm = g[c], bt = be[c];
  const float* xb = x + ((size_t)bn * 4) * C + c;
  float best = -1e30f;
  for (int kk = 0; kk < 4; ++kk) {
    float v = (xb[(size_t)kk * C] - mu) * inv * gm + bt;
    v = v > 0.f ? v : 0.2f * v;
    best = fmaxf(best, v);
  }
  out[t] = best;
}

__global__ void k_build_fl0(const float* __restrict__ pts, const float* __restrict__ cls,
                            float* __restrict__ out, long total)   // total = B*N
{
  long t = (long)blockIdx.x * 256 + threadIdx.x;
  if (t >= total) return;
  int b = (int)(t / NN);
  float* o = out + (size_t)t * 5;
  o[0] = cls[b * 2]; o[1] = cls[b * 2 + 1];
  o[2] = pts[(size_t)t * 3]; o[3] = pts[(size_t)t * 3 + 1]; o[4] = pts[(size_t)t * 3 + 2];
}

__global__ void k_transpose_out(const float* __restrict__ lg, float* __restrict__ out, long total)
{
  long t = (long)blockIdx.x * 256 + threadIdx.x;       // over B*NCLS*N
  if (t >= total) return;
  int n = (int)(t % NN);
  long bc = t / NN;
  int c = (int)(bc % NCLS);
  int b = (int)(bc / NCLS);
  out[t] = lg[((size_t)b * NN + n) * NCLS + c];
}

// =====================================================================
// Host orchestration
// =====================================================================
static inline unsigned gdim(long total) { return (unsigned)((total + 255) / 256); }

static void gemm(hipStream_t s, const float* A, const float* W, const float* bias,
                 const float* g, const float* bb, const float* res, float* C,
                 int M, int K, int N, int act)
{
  dim3 grid((N + 63) / 64, (M + 63) / 64);
  k_gemm_wmma<<<grid, 128, 0, s>>>(A, W, bias, g, bb, res, C, M, K, N, act);
}

extern "C" void kernel_launch(void* const* d_in, const int* in_sizes, int n_in,
                              void* d_out, int out_size, void* d_ws, size_t ws_size,
                              hipStream_t stream)
{
  const float* pts = (const float*)d_in[0];
  const float* cls = (const float*)d_in[1];

  // ---- params: JAX pytree leaf order (dict keys sorted, recursively) ----
  int q = 0;
  auto nxt = [&]() { return (const float*)d_in[2 + q++]; };
  struct Blk { const float *fc1_b,*fc1_w,*fc2_b,*fc2_w,*ln1_b,*ln1_g,*ln2_b,*ln2_g,*proj_b,*proj_w,*qkv_w; } blk[12];
  for (int i = 0; i < 12; ++i) {
    blk[i].fc1_b = nxt(); blk[i].fc1_w = nxt(); blk[i].fc2_b = nxt(); blk[i].fc2_w = nxt();
    blk[i].ln1_b = nxt(); blk[i].ln1_g = nxt(); blk[i].ln2_b = nxt(); blk[i].ln2_g = nxt();
    blk[i].proj_b = nxt(); blk[i].proj_w = nxt(); blk[i].qkv_w = nxt();
  }
  struct DG { const float *be1,*be2,*g1,*g2,*w1,*w2; } dg1p, dg2p;
  dg1p.be1 = nxt(); dg1p.be2 = nxt(); dg1p.g1 = nxt(); dg1p.g2 = nxt(); dg1p.w1 = nxt(); dg1p.w2 = nxt();
  dg2p.be1 = nxt(); dg2p.be2 = nxt(); dg2p.g1 = nxt(); dg2p.g2 = nxt(); dg2p.w1 = nxt(); dg2p.w2 = nxt();
  const float* e_b1 = nxt(); const float* e_b2 = nxt(); const float* e_b3 = nxt(); const float* e_b4 = nxt();
  const float* e_be1 = nxt(); const float* e_be3 = nxt(); const float* e_g1 = nxt(); const float* e_g3 = nxt();
  const float* e_w1 = nxt(); const float* e_w2 = nxt(); const float* e_w3 = nxt(); const float* e_w4 = nxt();
  struct FP { const float *be1,*be2,*g1,*g2,*w1,*w2; } fp0p, fp1p, fp2p;
  fp0p.be1 = nxt(); fp0p.be2 = nxt(); fp0p.g1 = nxt(); fp0p.g2 = nxt(); fp0p.w1 = nxt(); fp0p.w2 = nxt();
  fp1p.be1 = nxt(); fp1p.be2 = nxt(); fp1p.g1 = nxt(); fp1p.g2 = nxt(); fp1p.w1 = nxt(); fp1p.w2 = nxt();
  fp2p.be1 = nxt(); fp2p.be2 = nxt(); fp2p.g1 = nxt(); fp2p.g2 = nxt(); fp2p.w1 = nxt(); fp2p.w2 = nxt();
  const float* norm_b = nxt(); const float* norm_g = nxt();
  const float* pe_b1 = nxt(); const float* pe_b2 = nxt(); const float* pe_w1 = nxt(); const float* pe_w2 = nxt();
  const float* rd_b = nxt(); const float* rd_w = nxt();
  const float* sh_b1 = nxt(); const float* sh_b2 = nxt(); const float* sh_be1 = nxt();
  const float* sh_g1 = nxt(); const float* sh_w1 = nxt(); const float* sh_w2 = nxt();
  (void)n_in; (void)in_sizes; (void)out_size; (void)ws_size;

  // ---- workspace bump allocator ----
  size_t off = 0;
  auto alloc  = [&](size_t n) { float* p = (float*)((char*)d_ws + off); off += ((n * 4 + 255) & ~(size_t)255); return p; };
  auto alloci = [&](size_t n) { int*   p = (int*)  ((char*)d_ws + off); off += ((n * 4 + 255) & ~(size_t)255); return p; };

  const long RG  = (long)BB * GG;        // 2048 token rows
  const long RGM = RG * MM;              // 65536 encoder rows

  // persistent
  float* fxyz   = alloc((size_t)BB * 512 * 3);
  float* center = alloc((size_t)BB * 128 * 3);
  float* cp1    = alloc((size_t)BB * 256 * 3);
  int*   nidx   = alloci(RGM);
  float* nb     = alloc((size_t)RGM * 3);
  float* posh   = alloc(RG * 128);
  float* pos    = alloc(RG * DD);
  float* xA     = alloc(RG * DD);
  float* xB     = alloc(RG * DD);
  float* xin    = alloc(RG * DD);
  float* hln    = alloc(RG * DD);
  float* qkvb   = alloc(RG * 3 * DD);
  float* attn_o = alloc(RG * DD);
  float* x1     = alloc(RG * DD);
  float* f1b    = alloc(RG * 4 * DD);
  float* feats[3], *inter[3];
  for (int j = 0; j < 3; ++j) { feats[j] = alloc(RG * DD); inter[j] = alloc(RG * DD); }
  float* fl0base = alloc((size_t)BB * NN * 5);
  float* fl1  = alloc((size_t)BB * 512 * DD);
  float* fl2  = alloc((size_t)BB * 256 * DD);
  float* fl2b = alloc((size_t)BB * 256 * DD);
  float* fl1b = alloc((size_t)BB * 512 * DD);
  float* fl0  = alloc((size_t)BB * NN * DD);
  float* statsb = alloc(BB * 4 * 2);
  int*   kidx   = alloci((size_t)BB * NN * 4);
  float* kdist  = alloc((size_t)BB * NN * 3);
  const size_t mark = off;

  // ================= stage 0: FPS / kNN / grouping =================
  k_fps<<<BB, 256, 0, stream>>>(pts, fxyz, 512);
  k_subset3<<<gdim((long)BB * 128 * 3), 256, 0, stream>>>(fxyz, center, 512, (long)BB * 128 * 3);
  k_subset3<<<gdim((long)BB * 256 * 3), 256, 0, stream>>>(fxyz, cp1, 512, (long)BB * 256 * 3);
  float* cp0 = fxyz;  // (B,512,3)

  k_knn<32><<<dim3(2, BB), 64, 0, stream>>>(center, pts, nidx, nullptr, GG, NN);
  k_gather_nb<<<gdim(RGM), 256, 0, stream>>>(pts, center, nidx, nb, RGM);

  // ================= stage 1: encoder =================
  off = mark;
  float* h1     = alloc((size_t)RGM * 128);
  float* h2     = alloc((size_t)RGM * 256);
  float* hg     = alloc(RG * 256);
  float* hcat   = alloc((size_t)RGM * 512);
  float* h3     = alloc((size_t)RGM * 512);
  float* h4     = alloc((size_t)RGM * 256);
  float* tokenc = alloc(RG * 256);
  gemm(stream, nb, e_w1, e_b1, e_g1, e_be1, nullptr, h1, (int)RGM, 3, 128, 1);
  gemm(stream, h1, e_w2, e_b2, nullptr, nullptr, nullptr, h2, (int)RGM, 128, 256, 0);
  k_maxpool<<<gdim(RG * 256), 256, 0, stream>>>(h2, hg, MM, 256, RG * 256);
  k_concat_enc<<<gdim(RGM * 512), 256, 0, stream>>>(hg, h2, hcat, RGM * 512);
  gemm(stream, hcat, e_w3, e_b3, e_g3, e_be3, nullptr, h3, (int)RGM, 512, 512, 1);
  gemm(stream, h3, e_w4, e_b4, nullptr, nullptr, nullptr, h4, (int)RGM, 512, 256, 0);
  k_maxpool<<<gdim(RG * 256), 256, 0, stream>>>(h4, tokenc, MM, 256, RG * 256);
  gemm(stream, tokenc, rd_w, rd_b, nullptr, nullptr, nullptr, xA, (int)RG, ENCC, DD, 0);

  // positional embedding
  gemm(stream, center, pe_w1, pe_b1, nullptr, nullptr, nullptr, posh, (int)RG, 3, 128, 2);
  gemm(stream, posh, pe_w2, pe_b2, nullptr, nullptr, nullptr, pos, (int)RG, 128, DD, 0);

  // ================= stage 2: transformer blocks =================
  float* xcur = xA; float* xalt = xB;
  int fj = 0;
  for (int i = 0; i < 12; ++i) {
    const Blk& P = blk[i];
    k_add<<<gdim(RG * DD), 256, 0, stream>>>(xcur, pos, xin, RG * DD);
    k_ln<<<(unsigned)RG, 128, 0, stream>>>(xin, P.ln1_g, P.ln1_b, hln, DD);
    gemm(stream, hln, P.qkv_w, nullptr, nullptr, nullptr, nullptr, qkvb, (int)RG, DD, 3 * DD, 0);
    k_attn<<<BB * HH, 128, GG * DHh * 2 * sizeof(float), stream>>>(qkvb, attn_o);
    gemm(stream, attn_o, P.proj_w, P.proj_b, nullptr, nullptr, xin, x1, (int)RG, DD, DD, 0);
    k_ln<<<(unsigned)RG, 128, 0, stream>>>(x1, P.ln2_g, P.ln2_b, hln, DD);
    gemm(stream, hln, P.fc1_w, P.fc1_b, nullptr, nullptr, nullptr, f1b, (int)RG, DD, 4 * DD, 2);
    gemm(stream, f1b, P.fc2_w, P.fc2_b, nullptr, nullptr, x1, xalt, (int)RG, 4 * DD, DD, 0);
    float* tmp = xcur; xcur = xalt; xalt = tmp;
    if (i == 3 || i == 7 || i == 11) {
      (void)hipMemcpyAsync(feats[fj], xcur, RG * DD * sizeof(float), hipMemcpyDeviceToDevice, stream);
      ++fj;
    }
  }
  for (int j = 0; j < 3; ++j)
    k_ln<<<(unsigned)RG, 128, 0, stream>>>(feats[j], norm_g, norm_b, inter[j], DD);
  float* fl3 = inter[2];  // (B,128,384) at centers

  k_build_fl0<<<gdim((long)BB * NN), 256, 0, stream>>>(pts, cls, fl0base, (long)BB * NN);

  // ================= stage 3: feature propagation fp2 / fp1 =================
  off = mark;
  {
    float* fphc = alloc((size_t)BB * 256 * 387);
    float* fpm  = alloc((size_t)BB * 256 * 1536);
    k_knn<3><<<dim3(4, BB), 64, 0, stream>>>(cp1, center, kidx, kdist, 256, 128);
    k_fp_interp<<<gdim((long)BB * 256), 256, 0, stream>>>(kidx, kdist, inter[1], cp1, fphc, 256, 128, DD, 3, (long)BB * 256);
    gemm(stream, fphc, fp2p.w1, nullptr, fp2p.g1, fp2p.be1, nullptr, fpm, BB * 256, 387, 1536, 1);
    gemm(stream, fpm, fp2p.w2, nullptr, fp2p.g2, fp2p.be2, nullptr, fl2, BB * 256, 1536, DD, 1);
  }
  off = mark;
  {
    float* fphc = alloc((size_t)BB * 512 * 387);
    float* fpm  = alloc((size_t)BB * 512 * 1536);
    k_knn<3><<<dim3(8, BB), 64, 0, stream>>>(cp0, center, kidx, kdist, 512, 128);
    k_fp_interp<<<gdim((long)BB * 512), 256, 0, stream>>>(kidx, kdist, inter[0], cp0, fphc, 512, 128, DD, 3, (long)BB * 512);
    gemm(stream, fphc, fp1p.w1, nullptr, fp1p.g1, fp1p.be1, nullptr, fpm, BB * 512, 387, 1536, 1);
    gemm(stream, fpm, fp1p.w2, nullptr, fp1p.g2, fp1p.be2, nullptr, fl1, BB * 512, 1536, DD, 1);
  }

  // ================= stage 4: DGCNN dg2 (center/fl3 -> cp1/fl2) =================
  off = mark;
  {
    const int Q = 256, Kn = 128;
    long rows = (long)BB * Q * 4;
    float* gf  = alloc((size_t)rows * 768);
    float* gh  = alloc((size_t)rows * 512);
    float* ghm = alloc((size_t)BB * Q * 512);
    float* gf2 = alloc((size_t)rows * 1024);
    float* gh2 = alloc((size_t)rows * 384);
    k_knn<4><<<dim3(4, BB), 64, 0, stream>>>(cp1, center, kidx, nullptr, Q, Kn);
    k_graphfeat<<<gdim(rows), 256, 0, stream>>>(kidx, fl3, fl2, gf, Q, Kn, DD, rows);
    gemm(stream, gf, dg2p.w1, nullptr, nullptr, nullptr, nullptr, gh, (int)rows, 768, 512, 0);
    k_gn_stats<<<dim3(4, BB), 256, 0, stream>>>(gh, statsb, Q * 4, 512);
    k_gn_apply_max<<<gdim((long)BB * Q * 512), 256, 0, stream>>>(gh, statsb, dg2p.g1, dg2p.be1, ghm, Q, 512, (long)BB * Q * 512);
    k_knn<4><<<dim3(4, BB), 64, 0, stream>>>(cp1, cp1, kidx, nullptr, Q, Q);
    k_graphfeat<<<gdim(rows), 256, 0, stream>>>(kidx, ghm, ghm, gf2, Q, Q, 512, rows);
    gemm(stream, gf2, dg2p.w2, nullptr, nullptr, nullptr, nullptr, gh2, (int)rows, 1024, 384, 0);
    k_gn_stats<<<dim3(4, BB), 256, 0, stream>>>(gh2, statsb, Q * 4, 384);
    k_gn_apply_max<<<gdim((long)BB * Q * 384), 256, 0, stream>>>(gh2, statsb, dg2p.g2, dg2p.be2, fl2b, Q, 384, (long)BB * Q * 384);
  }

  // ================= stage 5: DGCNN dg1 (cp1/fl2b -> cp0/fl1) =================
  off = mark;
  {
    const int Q = 512, Kn = 256;
    long rows = (long)BB * Q * 4;
    float* gf  = alloc((size_t)rows * 768);
    float* gh  = alloc((size_t)rows * 512);
    float* ghm = alloc((size_t)BB * Q * 512);
    float* gf2 = alloc((size_t)rows * 1024);
    float* gh2 = alloc((size_t)rows * 384);
    k_knn<4><<<dim3(8, BB), 64, 0, stream>>>(cp0, cp1, kidx, nullptr, Q, Kn);
    k_graphfeat<<<gdim(rows), 256, 0, stream>>>(kidx, fl2b, fl1, gf, Q, Kn, DD, rows);
    gemm(stream, gf, dg1p.w1, nullptr, nullptr, nullptr, nullptr, gh, (int)rows, 768, 512, 0);
    k_gn_stats<<<dim3(4, BB), 256, 0, stream>>>(gh, statsb, Q * 4, 512);
    k_gn_apply_max<<<gdim((long)BB * Q * 512), 256, 0, stream>>>(gh, statsb, dg1p.g1, dg1p.be1, ghm, Q, 512, (long)BB * Q * 512);
    k_knn<4><<<dim3(8, BB), 64, 0, stream>>>(cp0, cp0, kidx, nullptr, Q, Q);
    k_graphfeat<<<gdim(rows), 256, 0, stream>>>(kidx, ghm, ghm, gf2, Q, Q, 512, rows);
    gemm(stream, gf2, dg1p.w2, nullptr, nullptr, nullptr, nullptr, gh2, (int)rows, 1024, 384, 0);
    k_gn_stats<<<dim3(4, BB), 256, 0, stream>>>(gh2, statsb, Q * 4, 384);
    k_gn_apply_max<<<gdim((long)BB * Q * 384), 256, 0, stream>>>(gh2, statsb, dg1p.g2, dg1p.be2, fl1b, Q, 384, (long)BB * Q * 384);
  }

  // ================= stage 6: fp0 + segmentation head =================
  off = mark;
  {
    long rows = (long)BB * NN;
    float* fphc  = alloc((size_t)rows * 389);
    float* fpm   = alloc((size_t)rows * 1536);
    float* headh = alloc((size_t)rows * 128);
    float* lgts  = alloc((size_t)rows * NCLS);
    k_knn<3><<<dim3(32, BB), 64, 0, stream>>>(pts, cp0, kidx, kdist, NN, 512);
    k_fp_interp<<<gdim(rows), 256, 0, stream>>>(kidx, kdist, fl1b, fl0base, fphc, NN, 512, DD, 5, rows);
    gemm(stream, fphc, fp0p.w1, nullptr, fp0p.g1, fp0p.be1, nullptr, fpm, (int)rows, 389, 1536, 1);
    gemm(stream, fpm, fp0p.w2, nullptr, fp0p.g2, fp0p.be2, nullptr, fl0, (int)rows, 1536, DD, 1);
    gemm(stream, fl0, sh_w1, sh_b1, sh_g1, sh_be1, nullptr, headh, (int)rows, DD, 128, 0);
    gemm(stream, headh, sh_w2, sh_b2, nullptr, nullptr, nullptr, lgts, (int)rows, 128, NCLS, 0);
    k_transpose_out<<<gdim((long)BB * NCLS * NN), 256, 0, stream>>>(lgts, (float*)d_out, (long)BB * NCLS * NN);
  }
}
